// MBaseBlock_58368605552723
// MI455X (gfx1250) — compile-verified, matrix operand order checked by reading, not run on hardware
//
#include <hip/hip_runtime.h>
#include <hip/hip_bf16.h>

// ---- CDNA5 WMMA types -------------------------------------------------------
typedef __bf16 bf16;
typedef __bf16 v16bf __attribute__((ext_vector_type(16)));
typedef float  v8f   __attribute__((ext_vector_type(8)));

#define BM 128
#define BN 64
#define BK 32
#define LS 40    // LDS row stride (bf16): 80B -> 16B-aligned b128 chunks

__device__ __forceinline__ float silu_f(float v) {
    return v * (1.0f / (1.0f + __expf(-v)));
}

// K-pair base for the 16-bit A/B fragment layout (ISA 7.12.2):
// lanes 0-15 hold K {0..7,16..23}, lanes 16-31 hold K {8..15,24..31}.
__device__ __forceinline__ int kpair_base(int v, int g) {
    return ((v < 4) ? (2 * v) : (16 + 2 * (v - 4))) + 8 * g;
}

// One 16B lane-chunk: global -> LDS via the CDNA5 async DMA path (ASYNCcnt).
// vdst = per-lane LDS byte address, vaddr = per-lane 64-bit global address.
__device__ __forceinline__ void async_copy_b128(unsigned lds_off, const void* gptr) {
    asm volatile("global_load_async_to_lds_b128 %0, %1, off"
                 :: "v"(lds_off), "v"(gptr)
                 : "memory");
}

__device__ __forceinline__ void wait_asynccnt0() {
    asm volatile("s_wait_asynccnt 0x0" ::: "memory");
}

// Issue async copies for one K-slice: A tile BMxBK (512 x 16B chunks, 2/thread),
// Bt tile BNxBK (256 chunks, 1/thread). Both bf16, K-contiguous rows.
__device__ __forceinline__ void stage_async(const bf16* __restrict__ A,
                                            const bf16* __restrict__ Bt,
                                            bf16* sA, bf16* sB,
                                            int tid, int m0, int n0, int K,
                                            int k0) {
    #pragma unroll
    for (int j = 0; j < 2; ++j) {
        int c = tid + 256 * j, r = c >> 2, q = c & 3;
        const void* g = (const void*)(A + (size_t)(m0 + r) * K + k0 + q * 8);
        unsigned    l = (unsigned)(size_t)(sA + r * LS + q * 8);
        async_copy_b128(l, g);
    }
    int r = tid >> 2, q = tid & 3;
    const void* g = (const void*)(Bt + (size_t)(n0 + r) * K + k0 + q * 8);
    unsigned    l = (unsigned)(size_t)(sB + r * LS + q * 8);
    async_copy_b128(l, g);
}

// C = A[MxK](bf16) * Bt[NxK](bf16)^T  (+bias)(silu)(+res), f32 accumulate.
// Double-buffered LDS fed by async-to-LDS DMA; one barrier per K-step.
template <bool BIAS, bool SILU, bool RES, bool RESF32, bool OUTF32>
__global__ __launch_bounds__(256) void gemm_bf16_wmma(
    const bf16* __restrict__ A, const bf16* __restrict__ Bt,
    const float* __restrict__ bias, const void* __restrict__ resv,
    void* __restrict__ Cv, int M, int N, int K) {
    __shared__ bf16 sA[2][BM * LS];   // [m][k]
    __shared__ bf16 sB[2][BN * LS];   // [n][k]

    const int tid  = threadIdx.x;
    const int lane = tid & 31;
    const int wid  = tid >> 5;                 // 8 waves: 4(M) x 2(N)
    const int m0   = blockIdx.y * BM;
    const int n0   = blockIdx.x * BN;
    const int wm   = (wid >> 1) * 32;
    const int wn   = (wid & 1) * 32;
    const int lr   = lane & 15;
    const int g    = lane >> 4;

    v8f acc[2][2] = {};

    stage_async(A, Bt, &sA[0][0], &sB[0][0], tid, m0, n0, K, 0);
    wait_asynccnt0();
    __syncthreads();

    const int nk = K >> 5;
    for (int ks = 0; ks < nk; ++ks) {
        const int cur = ks & 1;
        const bool hasN = (ks + 1) < nk;
        if (hasN)
            stage_async(A, Bt, &sA[cur ^ 1][0], &sB[cur ^ 1][0],
                        tid, m0, n0, K, (ks + 1) << 5);

        v16bf fa[2], fb[2];
        #pragma unroll
        for (int mi = 0; mi < 2; ++mi) {
            const int row = wm + mi * 16 + lr;
            #pragma unroll
            for (int v = 0; v < 8; ++v) {
                const int kb = kpair_base(v, g);
                fa[mi][2 * v]     = sA[cur][row * LS + kb];
                fa[mi][2 * v + 1] = sA[cur][row * LS + kb + 1];
            }
        }
        #pragma unroll
        for (int ni = 0; ni < 2; ++ni) {
            const int col = wn + ni * 16 + lr;
            #pragma unroll
            for (int v = 0; v < 8; ++v) {
                const int kb = kpair_base(v, g);
                fb[ni][2 * v]     = sB[cur][col * LS + kb];
                fb[ni][2 * v + 1] = sB[cur][col * LS + kb + 1];
            }
        }
        #pragma unroll
        for (int mi = 0; mi < 2; ++mi)
            #pragma unroll
            for (int ni = 0; ni < 2; ++ni)
                acc[mi][ni] = __builtin_amdgcn_wmma_f32_16x16x32_bf16(
                    false, fa[mi], false, fb[ni], (short)0, acc[mi][ni],
                    false, false);

        if (hasN) wait_asynccnt0();
        __syncthreads();
    }

    // epilogue: C/D layout M = r + 8*(lane/16), N = lane%16
    #pragma unroll
    for (int mi = 0; mi < 2; ++mi) {
        #pragma unroll
        for (int ni = 0; ni < 2; ++ni) {
            #pragma unroll
            for (int r = 0; r < 8; ++r) {
                const int row = m0 + wm + mi * 16 + g * 8 + r;
                const int col = n0 + wn + ni * 16 + lr;
                const size_t idx = (size_t)row * N + col;
                float v = acc[mi][ni][r];
                if (BIAS) v += bias[col];
                if (SILU) v = silu_f(v);
                if (RES)
                    v += RESF32 ? ((const float*)resv)[idx]
                                : (float)((const bf16*)resv)[idx];
                if (OUTF32) ((float*)Cv)[idx] = v;
                else        ((bf16*)Cv)[idx]  = (bf16)v;
            }
        }
    }
}

// fp32 -> bf16 flat convert
__global__ __launch_bounds__(256) void cvt_f32_bf16(const float* __restrict__ in,
                                                    bf16* __restrict__ out, int n) {
    int i = blockIdx.x * 256 + threadIdx.x;
    if (i < n) out[i] = (bf16)in[i];
}

// W[K][N] fp32 -> Wt[N][K] bf16
__global__ __launch_bounds__(256) void transpose_cvt(const float* __restrict__ W,
                                                     bf16* __restrict__ Wt,
                                                     int K, int N) {
    int i = blockIdx.x * 256 + threadIdx.x;
    if (i < N * K) {
        int n = i / K, k = i % K;
        Wt[i] = (bf16)W[(size_t)k * N + n];
    }
}

// Depthwise 3x3 conv (SAME) on x1 = xz[:, :384] + bias + SiLU, gated with
// silu(z = xz[:, 384:]) and the 4x cross-scan-collapse factor. bf16 in/out.
__global__ __launch_bounds__(256) void conv_gate_kernel(
    const bf16* __restrict__ xz, const float* __restrict__ cw,
    const float* __restrict__ cb, bf16* __restrict__ yg) {
    const int CI_ = 384, TW = 768, W_ = 64, H_ = 64;
    int t = blockIdx.x * 256 + threadIdx.x;    // exactly 32768*384 threads
    int c = t % CI_;
    int s = t / CI_;
    int w = s % W_;
    int h = (s / W_) % H_;
    int b = s / (W_ * H_);
    float acc = cb[c];
    #pragma unroll
    for (int dy = -1; dy <= 1; ++dy) {
        int y = h + dy;
        if (y < 0 || y >= H_) continue;
        #pragma unroll
        for (int dx = -1; dx <= 1; ++dx) {
            int x = w + dx;
            if (x < 0 || x >= W_) continue;
            acc += (float)xz[(size_t)((b * H_ + y) * W_ + x) * TW + c] *
                   cw[c * 9 + (dy + 1) * 3 + (dx + 1)];
        }
    }
    float z = (float)xz[(size_t)s * TW + CI_ + c];
    yg[(size_t)s * CI_ + c] = (bf16)(4.0f * silu_f(acc) * silu_f(z));
}

extern "C" void kernel_launch(void* const* d_in, const int* in_sizes, int n_in,
                              void* d_out, int out_size, void* d_ws,
                              size_t ws_size, hipStream_t stream) {
    const float* x    = (const float*)d_in[0];   // [8,64,64,192]
    const float* Win  = (const float*)d_in[1];   // [192,768]
    const float* cw   = (const float*)d_in[2];   // [384,1,3,3]
    const float* cb   = (const float*)d_in[3];   // [384]
    const float* Wout = (const float*)d_in[4];   // [384,192]
    const float* Wm1  = (const float*)d_in[5];   // [192,768]
    const float* bm1  = (const float*)d_in[6];   // [768]
    const float* Wm2  = (const float*)d_in[7];   // [768,192]
    const float* bm2  = (const float*)d_in[8];   // [192]
    float* out = (float*)d_out;                  // [8,64,64,192] fp32

    const int M = 8 * 64 * 64;                   // 32768
    const int C = 192, CI = 384, CI2 = 768;

    // workspace layout (bf16 intermediates; all offsets 16B aligned)
    char* ws = (char*)d_ws;
    size_t off = 0;
    bf16* xb    = (bf16*)(ws + off); off += (size_t)M * C  * 2;   // x in bf16
    bf16* xz    = (bf16*)(ws + off); off += (size_t)M * CI2 * 2;  // xz / h
    bf16* yg    = (bf16*)(ws + off); off += (size_t)M * CI * 2;   // gated conv out
    bf16* outb  = (bf16*)(ws + off); off += (size_t)M * C  * 2;   // out (bf16)
    bf16* WinT  = (bf16*)(ws + off); off += (size_t)CI2 * C * 2;  // [768][192]
    bf16* WoutT = (bf16*)(ws + off); off += (size_t)C * CI * 2;   // [192][384]
    bf16* Wm1T  = (bf16*)(ws + off); off += (size_t)CI2 * C * 2;  // [768][192]
    bf16* Wm2T  = (bf16*)(ws + off); off += (size_t)C * CI2 * 2;  // [192][768]

    dim3 blk(256);

    // ---- prep: convert x, transpose+convert weights -------------------------
    cvt_f32_bf16<<<dim3((M * C + 255) / 256), blk, 0, stream>>>(x, xb, M * C);
    transpose_cvt<<<dim3((CI2 * C + 255) / 256), blk, 0, stream>>>(Win,  WinT,  C,  CI2);
    transpose_cvt<<<dim3((C * CI  + 255) / 256), blk, 0, stream>>>(Wout, WoutT, CI, C);
    transpose_cvt<<<dim3((CI2 * C + 255) / 256), blk, 0, stream>>>(Wm1,  Wm1T,  C,  CI2);
    transpose_cvt<<<dim3((C * CI2 + 255) / 256), blk, 0, stream>>>(Wm2,  Wm2T,  CI2, C);

    // 1) xz = x @ Win                                   -> bf16
    gemm_bf16_wmma<false, false, false, false, false>
        <<<dim3(CI2 / BN, M / BM), blk, 0, stream>>>(
            xb, WinT, nullptr, nullptr, xz, M, CI2, C);

    // 2) depthwise conv + silu, gate, x4                -> bf16
    conv_gate_kernel<<<dim3((M * CI) / 256), blk, 0, stream>>>(xz, cw, cb, yg);

    // 3) out = x + yg @ Wout        (residual fp32 x)   -> bf16
    gemm_bf16_wmma<false, false, true, true, false>
        <<<dim3(C / BN, M / BM), blk, 0, stream>>>(
            yg, WoutT, nullptr, x, outb, M, C, CI);

    // 4) h = silu(out @ Wm1 + bm1)  (reuse xz buffer)   -> bf16
    gemm_bf16_wmma<true, true, false, false, false>
        <<<dim3(CI2 / BN, M / BM), blk, 0, stream>>>(
            outb, Wm1T, bm1, nullptr, xz, M, CI2, C);

    // 5) final = out + h @ Wm2 + bm2 (residual bf16 out) -> fp32 d_out
    gemm_bf16_wmma<true, false, true, false, true>
        <<<dim3(C / BN, M / BM), blk, 0, stream>>>(
            xz, Wm2T, bm2, outb, out, M, C, CI2);
}
